// GQABlock_76235669504334
// MI455X (gfx1250) — compile-verified, matrix operand order checked by reading, not run on hardware
//
#include <hip/hip_runtime.h>
#include <hip/hip_bf16.h>

// ---------------------------------------------------------------------------
// GQA block for MI455X (gfx1250, wave32, WMMA).
// All matmuls via v_wmma_f32_16x16x32_bf16 (bf16 operands, f32 accumulate).
// LDS tiles are PAIR-PACKED so each WMMA fragment load is two ds_load_b128.
// Flash attention: double-buffered K/V tiles with register-staged pipelining
// + global_prefetch of the tile after next; the S x S score matrix never
// touches HBM. All staging loops are constant-trip-count so global loads
// batch under LOADcnt instead of load->wait->store serialization.
// ---------------------------------------------------------------------------

typedef __bf16  bf16_t;
typedef __bf16  v16bf __attribute__((ext_vector_type(16)));
typedef float   v8f   __attribute__((ext_vector_type(8)));

#define NUM_HEADS    16
#define NUM_KV_HEADS 4
#define HEAD_DIM     64
#define SEQ          2048
#define DMODEL       1024

static __device__ __forceinline__ v8f wmma_bf16(v16bf a, v16bf b, v8f c) {
    return __builtin_amdgcn_wmma_f32_16x16x32_bf16(
        false, a, false, b, (short)0, c, false, false);
}

static __device__ __forceinline__ unsigned int pack_bf16(float a, float b) {
    union { bf16_t h[2]; unsigned int u; } x;
    x.h[0] = (bf16_t)a; x.h[1] = (bf16_t)b;
    return x.u;
}

static __device__ __forceinline__ unsigned short bf16_bits(float f) {
    union { bf16_t h; unsigned short s; } u; u.h = (bf16_t)f; return u.s;
}

// Fragment loader. `row` points at a pair-packed LDS row (dword kp holds
// bf16 k=2kp,2kp+1). Fragment dwords i=0..3 -> kp = t*16 + half*4 + i,
// i=4..7 -> kp = t*16 + 8 + half*4 + (i-4): two 128-bit LDS loads.
static __device__ __forceinline__ v16bf load_frag(const unsigned int* row,
                                                  int half, int t) {
    const unsigned int* p = row + t * 16 + half * 4;
    uint4 lo = *(const uint4*)(p);
    uint4 hi = *(const uint4*)(p + 8);
    union { unsigned int u[8]; v16bf v; } w;
    w.u[0] = lo.x; w.u[1] = lo.y; w.u[2] = lo.z; w.u[3] = lo.w;
    w.u[4] = hi.x; w.u[5] = hi.y; w.u[6] = hi.z; w.u[7] = hi.w;
    return w.v;
}

// ===========================================================================
// Kernel 1: out[b,h,s,dh](bf16) = rope( X(f32)[4096x1024] @ W(f32)[1024xN] )
// 128 threads (4 waves); block tile 64x64; wave w owns rows [16w,16w+16).
// ===========================================================================
__global__ void proj_rope_kernel(const float* __restrict__ X,
                                 const float* __restrict__ W,
                                 bf16_t* __restrict__ Out,
                                 int N, int do_rope,
                                 const int* __restrict__ posOff) {
    __shared__ unsigned int Apack[64][20];   // 16 dwords (k=0..31) + pad
    __shared__ unsigned int Bpack[64][20];

    const int tid  = threadIdx.x;
    const int wave = tid >> 5, lane = tid & 31;
    const int half = lane >> 4, hl = lane & 15;
    const int rowBlock = blockIdx.x * 64;
    const int colBlock = blockIdx.y * 64;
    const int Hout = N >> 6;

    v8f acc[4] = {};

    for (int k0 = 0; k0 < DMODEL; k0 += 32) {
        // --- load phase: batch all global loads, then convert+store ---
        float2 av[8];
        float  bw0[8], bw1[8];
#pragma unroll
        for (int n = 0; n < 8; ++n) {
            int i = tid + n * 128;                // A: r = i>>4, kp = i&15
            av[n] = *(const float2*)&X[(size_t)(rowBlock + (i >> 4)) * DMODEL +
                                       k0 + 2 * (i & 15)];
        }
#pragma unroll
        for (int n = 0; n < 8; ++n) {
            int i = tid + n * 128;                // B: n = i&63, kp = i>>6
            int nn = i & 63, kp = i >> 6;
            bw0[n] = W[(size_t)(k0 + 2 * kp) * N + colBlock + nn];
            bw1[n] = W[(size_t)(k0 + 2 * kp + 1) * N + colBlock + nn];
        }
        // L2 prefetch of the next k-slab (global_prefetch_b8)
        if (k0 + 32 < DMODEL) {
            __builtin_prefetch(&X[(size_t)(rowBlock + (tid >> 1)) * DMODEL +
                                  k0 + 32 + (tid & 1) * 16], 0, 1);
            __builtin_prefetch(&W[(size_t)(k0 + 32 + (tid >> 2)) * N +
                                  colBlock + (tid & 3) * 16], 0, 1);
        }
#pragma unroll
        for (int n = 0; n < 8; ++n) {
            int i = tid + n * 128;
            Apack[i >> 4][i & 15] = pack_bf16(av[n].x, av[n].y);
        }
#pragma unroll
        for (int n = 0; n < 8; ++n) {
            int i = tid + n * 128;
            Bpack[i & 63][i >> 6] = pack_bf16(bw0[n], bw1[n]);
        }
        __syncthreads();

        v16bf a = load_frag(&Apack[wave * 16 + hl][0], half, 0);
#pragma unroll
        for (int j = 0; j < 4; ++j) {
            v16bf b = load_frag(&Bpack[j * 16 + hl][0], half, 0);
            acc[j] = wmma_bf16(a, b, acc[j]);
        }
        __syncthreads();
    }

    // RoPE epilogue: column dh (<32) pairs with dh+32 -> acc[j] with acc[j+2].
    const float LOG_BASE_OVER_HALF = 0.28782313662425572f;   // ln(10000)/32
    const int off = *posOff;

#pragma unroll
    for (int j = 0; j < 2; ++j) {
        const int gc0 = colBlock + j * 16 + hl;   // dh in [0,32) within head
        const int h    = gc0 >> 6;
        const int irot = gc0 & 31;
        const float invf = __expf(-(float)irot * LOG_BASE_OVER_HALF);
#pragma unroll
        for (int r = 0; r < 8; ++r) {
            const int gr = rowBlock + wave * 16 + half * 8 + r;
            const int bb = gr >> 11, s = gr & (SEQ - 1);
            float x0 = acc[j][r], x1 = acc[j + 2][r];
            float o0 = x0, o1 = x1;
            if (do_rope) {
                float ang = (float)(off + s) * invf;
                float c = __cosf(ang), sn = __sinf(ang);
                o0 = x0 * c - x1 * sn;
                o1 = x1 * c + x0 * sn;
            }
            float o0p = __shfl_xor(o0, 1);
            float o1p = __shfl_xor(o1, 1);
            if ((hl & 1) == 0) {
                size_t base = (((size_t)bb * Hout + h) * SEQ + s) * HEAD_DIM;
                unsigned int* Od = (unsigned int*)Out;
                Od[(base + (gc0 & 63)) >> 1]      = pack_bf16(o0, o0p);
                Od[(base + (gc0 & 63) + 32) >> 1] = pack_bf16(o1, o1p);
            }
        }
    }
}

// ===========================================================================
// Kernel 2: flash attention. Grid (S/64, B*H); 128 threads (4 waves); each
// wave owns 16 query rows. Online softmax over causal 64-key tiles.
// Double-buffered K/V with register-staged pipelining + L2 prefetch.
// ===========================================================================
__global__ void flash_attn_kernel(const bf16_t* __restrict__ Q,
                                  const bf16_t* __restrict__ K,
                                  const bf16_t* __restrict__ V,
                                  bf16_t* __restrict__ OutAttn) {
    __shared__ unsigned int Qp[64][36];         // q-row, 32 dh-pair dwords
    __shared__ unsigned int Kp[2][64][36];      // key-row, 32 dh-pair dwords
    __shared__ unsigned int Vp[2][64][36];      // dh-row, 32 key-pair dwords
    __shared__ unsigned int Pp[4][16][36];      // per-wave P, q-row x key-pairs

    const int tid  = threadIdx.x;
    const int wave = tid >> 5, lane = tid & 31;
    const int half = lane >> 4, hl = lane & 15;
    const int qb = blockIdx.x;
    const int bh = blockIdx.y;
    const int b = bh >> 4, h = bh & 15;
    const int kvh = h >> 2;                     // GQA: 4 Q heads per KV head

    const size_t qbase  = (((size_t)b * NUM_HEADS + h) * SEQ + (size_t)qb * 64) * HEAD_DIM;
    const size_t kvbase = (((size_t)b * NUM_KV_HEADS + kvh) * SEQ) * HEAD_DIM;

    const unsigned int*   Kg0 = (const unsigned int*)(K + kvbase);
    const unsigned short* Vg0 = (const unsigned short*)(V + kvbase);

    // Stage Q tile (u32 copies of contiguous bf16 pairs), batched loads
    {
        const unsigned int* Qg = (const unsigned int*)(Q + qbase);
        unsigned int qreg[16];
#pragma unroll
        for (int n = 0; n < 16; ++n) qreg[n] = Qg[tid + n * 128];
#pragma unroll
        for (int n = 0; n < 16; ++n) {
            int i = tid + n * 128;
            Qp[i >> 5][i & 31] = qreg[n];
        }
    }

    // Stage K/V tile 0 into buffer 0 (V is transpose-packed over keys)
    unsigned int kreg[16], vreg[16];
#pragma unroll
    for (int n = 0; n < 16; ++n) {
        int i = tid + n * 128;
        kreg[n] = Kg0[i];
        int dh = i & 63, kp = i >> 6;
        vreg[n] = ((unsigned int)Vg0[(2 * kp + 1) * HEAD_DIM + dh] << 16) |
                  (unsigned int)Vg0[(2 * kp) * HEAD_DIM + dh];
    }
#pragma unroll
    for (int n = 0; n < 16; ++n) {
        int i = tid + n * 128;
        Kp[0][i >> 5][i & 31] = kreg[n];
        Vp[0][i & 63][i >> 6] = vreg[n];
    }
    __syncthreads();

    v16bf qf[2];
#pragma unroll
    for (int t = 0; t < 2; ++t)
        qf[t] = load_frag(&Qp[wave * 16 + hl][0], half, t);

    v8f o[4] = {};
    float mrow[8], lrow[8];
#pragma unroll
    for (int r = 0; r < 8; ++r) { mrow[r] = -1e30f; lrow[r] = 0.0f; }

    const float scale = 0.125f;   // 1/sqrt(64)

    for (int kt = 0; kt <= qb; ++kt) {
        const int buf = kt & 1;
        const bool more = kt < qb;

        // Issue global loads for tile kt+1 now; they complete under compute.
        if (more) {
            const unsigned int*   Kg = Kg0 + (size_t)(kt + 1) * 64 * 32;
            const unsigned short* Vg = Vg0 + (size_t)(kt + 1) * 64 * HEAD_DIM;
#pragma unroll
            for (int n = 0; n < 16; ++n) {
                int i = tid + n * 128;
                kreg[n] = Kg[i];
                int dh = i & 63, kp = i >> 6;
                vreg[n] = ((unsigned int)Vg[(2 * kp + 1) * HEAD_DIM + dh] << 16) |
                          (unsigned int)Vg[(2 * kp) * HEAD_DIM + dh];
            }
            if (kt + 2 <= qb) {   // L2 prefetch of the tile after next
                __builtin_prefetch(Kg0 + (size_t)(kt + 2) * 64 * 32 + tid * 16, 0, 1);
                __builtin_prefetch(Vg0 + (size_t)(kt + 2) * 64 * HEAD_DIM + tid * 32, 0, 1);
            }
        }

        // scores = Q @ K^T : B[k=dh][n=key] = K[key][dh] (pair-packed rows)
        v8f sc[4] = {};
#pragma unroll
        for (int j = 0; j < 4; ++j)
#pragma unroll
            for (int t = 0; t < 2; ++t) {
                v16bf bfr = load_frag(&Kp[buf][j * 16 + hl][0], half, t);
                sc[j] = wmma_bf16(qf[t], bfr, sc[j]);
            }

        // scale + causal mask + per-row max (row spans 16 lanes of a half)
        float tmax[8];
#pragma unroll
        for (int r = 0; r < 8; ++r) {
            const int qrow = qb * 64 + wave * 16 + half * 8 + r;
            float mx = -1e30f;
#pragma unroll
            for (int j = 0; j < 4; ++j) {
                float v = sc[j][r] * scale;
                int kcol = kt * 64 + j * 16 + hl;
                if (kcol > qrow) v = -1e30f;
                sc[j][r] = v;
                mx = fmaxf(mx, v);
            }
            tmax[r] = mx;
        }
#pragma unroll
        for (int d = 1; d < 16; d <<= 1)
#pragma unroll
            for (int r = 0; r < 8; ++r)
                tmax[r] = fmaxf(tmax[r], __shfl_xor(tmax[r], d, 32));

        // online softmax update
        float psum[8];
#pragma unroll
        for (int r = 0; r < 8; ++r) {
            float mnew = fmaxf(mrow[r], tmax[r]);
            float corr = __expf(mrow[r] - mnew);
            mrow[r] = mnew;
            lrow[r] *= corr;
#pragma unroll
            for (int j = 0; j < 4; ++j) o[j][r] *= corr;
            float ps = 0.0f;
#pragma unroll
            for (int j = 0; j < 4; ++j) {
                float p = __expf(sc[j][r] - mnew);
                sc[j][r] = p;
                ps += p;
            }
            psum[r] = ps;
        }
#pragma unroll
        for (int d = 1; d < 16; d <<= 1)
#pragma unroll
            for (int r = 0; r < 8; ++r)
                psum[r] += __shfl_xor(psum[r], d, 32);
#pragma unroll
        for (int r = 0; r < 8; ++r) lrow[r] += psum[r];

        // relayout P (acc layout) -> pair-packed A rows in wave-private LDS
#pragma unroll
        for (int j = 0; j < 4; ++j)
#pragma unroll
            for (int r = 0; r < 8; ++r) {
                unsigned short* prow = (unsigned short*)&Pp[wave][half * 8 + r][0];
                prow[j * 16 + hl] = bf16_bits(sc[j][r]);
            }

        // O += P @ V   (wave-private LDS RAW ordered by DScnt)
        v16bf pf[2];
#pragma unroll
        for (int t = 0; t < 2; ++t)
            pf[t] = load_frag(&Pp[wave][hl][0], half, t);
#pragma unroll
        for (int j = 0; j < 4; ++j)
#pragma unroll
            for (int t = 0; t < 2; ++t) {
                v16bf bfr = load_frag(&Vp[buf][j * 16 + hl][0], half, t);
                o[j] = wmma_bf16(pf[t], bfr, o[j]);
            }

        // commit staged registers to the other buffer for iteration kt+1
        __syncthreads();
        if (more) {
#pragma unroll
            for (int n = 0; n < 16; ++n) {
                int i = tid + n * 128;
                Kp[buf ^ 1][i >> 5][i & 31] = kreg[n];
                Vp[buf ^ 1][i & 63][i >> 6] = vreg[n];
            }
        }
        __syncthreads();
    }

    // normalize, pack lane pairs, store [B, S, H*64] bf16
#pragma unroll
    for (int r = 0; r < 8; ++r) {
        float linv = 1.0f / lrow[r];
        int s = qb * 64 + wave * 16 + half * 8 + r;
#pragma unroll
        for (int j = 0; j < 4; ++j) {
            float val = o[j][r] * linv;
            float oth = __shfl_xor(val, 1);
            if ((hl & 1) == 0) {
                size_t e = ((size_t)b * SEQ + s) * DMODEL + h * HEAD_DIM + j * 16 + hl;
                ((unsigned int*)OutAttn)[e >> 1] = pack_bf16(val, oth);
            }
        }
    }
}

// ===========================================================================
// Kernel 3: out(f32)[4096x1024] = Attn(bf16) @ Wo(f32 -> bf16)
// ===========================================================================
__global__ void out_proj_kernel(const bf16_t* __restrict__ A,
                                const float* __restrict__ W,
                                float* __restrict__ Out) {
    __shared__ unsigned int Apack[64][20];
    __shared__ unsigned int Bpack[64][20];

    const int tid  = threadIdx.x;
    const int wave = tid >> 5, lane = tid & 31;
    const int half = lane >> 4, hl = lane & 15;
    const int rowBlock = blockIdx.x * 64;
    const int colBlock = blockIdx.y * 64;

    v8f acc[4] = {};

    for (int k0 = 0; k0 < DMODEL; k0 += 32) {
        const unsigned int* Ag = (const unsigned int*)A;
        unsigned int areg[8];
        float bw0[8], bw1[8];
#pragma unroll
        for (int n = 0; n < 8; ++n) {
            int i = tid + n * 128;
            areg[n] = Ag[((size_t)(rowBlock + (i >> 4)) * DMODEL + k0) / 2 + (i & 15)];
        }
#pragma unroll
        for (int n = 0; n < 8; ++n) {
            int i = tid + n * 128;
            int nn = i & 63, kp = i >> 6;
            bw0[n] = W[(size_t)(k0 + 2 * kp) * DMODEL + colBlock + nn];
            bw1[n] = W[(size_t)(k0 + 2 * kp + 1) * DMODEL + colBlock + nn];
        }
        if (k0 + 32 < DMODEL) {
            __builtin_prefetch(A + (size_t)(rowBlock + (tid & 63)) * DMODEL + k0 + 32, 0, 1);
            __builtin_prefetch(&W[(size_t)(k0 + 32 + (tid >> 2)) * DMODEL +
                                  colBlock + (tid & 3) * 16], 0, 1);
        }
#pragma unroll
        for (int n = 0; n < 8; ++n) {
            int i = tid + n * 128;
            Apack[i >> 4][i & 15] = areg[n];
        }
#pragma unroll
        for (int n = 0; n < 8; ++n) {
            int i = tid + n * 128;
            Bpack[i & 63][i >> 6] = pack_bf16(bw0[n], bw1[n]);
        }
        __syncthreads();

        v16bf a = load_frag(&Apack[wave * 16 + hl][0], half, 0);
#pragma unroll
        for (int j = 0; j < 4; ++j) {
            v16bf b = load_frag(&Bpack[j * 16 + hl][0], half, 0);
            acc[j] = wmma_bf16(a, b, acc[j]);
        }
        __syncthreads();
    }

    // pack lane pairs -> float2 stores from even lanes
#pragma unroll
    for (int j = 0; j < 4; ++j)
#pragma unroll
        for (int r = 0; r < 8; ++r) {
            float val = acc[j][r];
            float oth = __shfl_xor(val, 1);
            if ((hl & 1) == 0) {
                int gr = rowBlock + wave * 16 + half * 8 + r;
                int gc = colBlock + j * 16 + hl;
                float2 v2; v2.x = val; v2.y = oth;
                *(float2*)&Out[(size_t)gr * DMODEL + gc] = v2;
            }
        }
}

// ===========================================================================
extern "C" void kernel_launch(void* const* d_in, const int* in_sizes, int n_in,
                              void* d_out, int out_size, void* d_ws, size_t ws_size,
                              hipStream_t stream) {
    (void)in_sizes; (void)n_in; (void)out_size; (void)ws_size;

    const float* x  = (const float*)d_in[0];
    const float* Wq = (const float*)d_in[1];
    const float* Wk = (const float*)d_in[2];
    const float* Wv = (const float*)d_in[3];
    const float* Wo = (const float*)d_in[4];
    const int* posOff = (const int*)d_in[5];

    const size_t QELEM  = (size_t)2 * NUM_HEADS * SEQ * HEAD_DIM;
    const size_t KVELEM = (size_t)2 * NUM_KV_HEADS * SEQ * HEAD_DIM;

    bf16_t* qws = (bf16_t*)d_ws;
    bf16_t* kws = qws + QELEM;
    bf16_t* vws = kws + KVELEM;
    bf16_t* aws = vws + KVELEM;   // attn output [B, S, 1024] bf16

    dim3 blk(128);
    proj_rope_kernel<<<dim3(64, 16), blk, 0, stream>>>(x, Wq, qws, DMODEL, 1, posOff);
    proj_rope_kernel<<<dim3(64, 4),  blk, 0, stream>>>(x, Wk, kws, NUM_KV_HEADS * HEAD_DIM, 1, posOff);
    proj_rope_kernel<<<dim3(64, 4),  blk, 0, stream>>>(x, Wv, vws, NUM_KV_HEADS * HEAD_DIM, 0, posOff);
    flash_attn_kernel<<<dim3(SEQ / 64, 2 * NUM_HEADS), blk, 0, stream>>>(qws, kws, vws, aws);
    out_proj_kernel<<<dim3(64, 16), blk, 0, stream>>>(aws, Wo, (float*)d_out);
}
